// InstructTimeAttention_4363686773164
// MI455X (gfx1250) — matrix operand order repaired and checked by reading, not compiled, not run on hardware
//
#include <hip/hip_runtime.h>
#include <math.h>

// B=8 H=8 Lq=64 Lp=512 V=32 DIM=512 HD=64 ; /SCALE == *8
typedef __attribute__((ext_vector_type(16))) __bf16 v16bf;
typedef __attribute__((ext_vector_type(8)))  __bf16 v8bf;
typedef __attribute__((ext_vector_type(8)))  float  v8f;

__device__ inline v8f wmma_bf16(v16bf a, v16bf b, v8f c) {
  // (neg_a, A, neg_b, B, c_mod, C, reuse_a, reuse_b)
  return __builtin_amdgcn_wmma_f32_16x16x32_bf16(false, a, false, b, (short)0, c, false, false);
}

// A-operand: lanes0-15 rows M, elems 0..7 = K[half*8 .. +8), elems 8..15 = K[16+half*8 ..)
__device__ inline v16bf frag_a(const __bf16* p, int half) {
  v8bf lo = *(const v8bf*)(p + half * 8);
  v8bf hi = *(const v8bf*)(p + 16 + half * 8);
  v16bf r;
#pragma unroll
  for (int j = 0; j < 8; ++j) { r[j] = lo[j]; r[j + 8] = hi[j]; }
  return r;
}
// B-operand: lane = column, elems 0..7 = K[half*8 .. +8), elems 8..15 = K[16+half*8 ..)
__device__ inline v16bf frag_b(const __bf16* p, int half) {
  v8bf lo = *(const v8bf*)(p + half * 8);
  v8bf hi = *(const v8bf*)(p + 16 + half * 8);
  v16bf r;
#pragma unroll
  for (int j = 0; j < 8; ++j) { r[j] = lo[j]; r[j + 8] = hi[j]; }
  return r;
}

// ---------------------------------------------------------------------------
// memMean[b,v,e] = mean over Lp of memory[b,lp,v,e]   (one streaming pass)
// ---------------------------------------------------------------------------
__global__ __launch_bounds__(256) void mean_kernel(const float* __restrict__ mem,
                                                   float* __restrict__ out) {
  int i = blockIdx.x * 256 + threadIdx.x;     // 0..131071 = ((b*32)+v)*512+e
  int e = i & 511;
  int v = (i >> 9) & 31;
  int b = i >> 14;
  const float* p = mem + (long)b * 8388608 + (long)v * 512 + e;
  float s = 0.f;
  for (int lp = 0; lp < 512; ++lp) s += p[(long)lp * 16384];
  out[i] = s * (1.0f / 512.0f);
}

// ---------------------------------------------------------------------------
// Generic batched GEMM  C = alpha * A @ B(^T) + bias, bf16 WMMA, f32 in/out.
// Block tile 64x64 (8 waves, each 16x32). All dims multiples of 32/64.
// z-batch decomposed as (b,h) with independent pointer strides.
// ---------------------------------------------------------------------------
__global__ __launch_bounds__(256) void gemm_wmma_kernel(
    const float* __restrict__ A, const float* __restrict__ Bm,
    float* __restrict__ C, const float* __restrict__ bias,
    int M, int N, int K, int lda, int ldb, int ldc,
    int transB, float alpha, int nh,
    long sAb, long sAh, long sBb, long sBh, long sCb, long sCh)
{
  __shared__ __align__(16) __bf16 Als[64 * 32];   // [row][k]
  __shared__ __align__(16) __bf16 Bls[64 * 32];   // [n][k] (k-contiguous)
  const int z = blockIdx.z;
  const int zb = z / nh, zh = z - zb * nh;
  A  += zb * sAb + zh * sAh;
  Bm += zb * sBb + zh * sBh;
  C  += zb * sCb + zh * sCh;
  const int row0 = blockIdx.y * 64;
  const int col0 = blockIdx.x * 64;
  const int t    = threadIdx.x;
  const int lane = t & 31, w = t >> 5;
  const int wm = w & 3, wn = w >> 2;          // wave grid 4 x 2
  const int half = lane >> 4, lr = lane & 15;
  v8f c0 = {}, c1 = {};

  const int a_row = t >> 2, a_k0 = (t & 3) * 8;
  const int b_kk  = t >> 3, b_n0 = (t & 7) * 8;

  for (int k0 = 0; k0 < K; k0 += 32) {
    __syncthreads();
    { // stage A tile 64x32 (coalesced along K), convert f32->bf16
      const float* ga = A + (long)(row0 + a_row) * lda + (k0 + a_k0);
      float4 f0 = *(const float4*)ga;
      float4 f1 = *(const float4*)(ga + 4);
      __bf16* d = &Als[a_row * 32 + a_k0];
      d[0]=(__bf16)f0.x; d[1]=(__bf16)f0.y; d[2]=(__bf16)f0.z; d[3]=(__bf16)f0.w;
      d[4]=(__bf16)f1.x; d[5]=(__bf16)f1.y; d[6]=(__bf16)f1.z; d[7]=(__bf16)f1.w;
    }
    { // stage B tile 32x64 into k-contiguous [n][k] layout
      if (!transB) {
        const float* gb = Bm + (long)(k0 + b_kk) * ldb + (col0 + b_n0);
        float4 f0 = *(const float4*)gb;
        float4 f1 = *(const float4*)(gb + 4);
        float v[8] = {f0.x, f0.y, f0.z, f0.w, f1.x, f1.y, f1.z, f1.w};
#pragma unroll
        for (int i = 0; i < 8; ++i) Bls[(b_n0 + i) * 32 + b_kk] = (__bf16)v[i];
      } else {
#pragma unroll
        for (int i = 0; i < 8; ++i)
          Bls[(b_n0 + i) * 32 + b_kk] =
              (__bf16)Bm[(long)(col0 + b_n0 + i) * ldb + (k0 + b_kk)];
      }
    }
    __syncthreads();
    v16bf a  = frag_a(&Als[(wm * 16 + lr) * 32], half);
    v16bf b0 = frag_b(&Bls[(wn * 32 + lr) * 32], half);
    v16bf b1 = frag_b(&Bls[(wn * 32 + 16 + lr) * 32], half);
    c0 = wmma_bf16(a, b0, c0);
    c1 = wmma_bf16(a, b1, c1);
  }

  // D layout: VGPR r -> row r (+8 for upper lane half), col = lane%16
  const int rbase = row0 + wm * 16 + half * 8;
  const int n0g = col0 + wn * 32 + lr;
  const int n1g = n0g + 16;
  const float add0 = bias ? bias[n0g] : 0.0f;
  const float add1 = bias ? bias[n1g] : 0.0f;
#pragma unroll
  for (int r = 0; r < 8; ++r) {
    int rr = rbase + r;
    C[(long)rr * ldc + n0g] = alpha * c0[r] + add0;
    C[(long)rr * ldc + n1g] = alpha * c1[r] + add1;
  }
}

// ---------------------------------------------------------------------------
// Stage-1 channel softmax -> q-summed weights  w[b,h,v]
// scores[q,v] = 8 * sum_d qW[b, h*8+q/8, (q%8)*64+d] * kcW[b, h*4+v/8, (v%8)*64+d]
// ---------------------------------------------------------------------------
__global__ __launch_bounds__(256) void ch_softmax_kernel(
    const float* __restrict__ qW, const float* __restrict__ kcW,
    float* __restrict__ wsum)
{
  __shared__ float sc[64][32];
  const int h = blockIdx.x & 7, b = blockIdx.x >> 3;
  const int t = threadIdx.x;
  const int q = t >> 2, vb = (t & 3) * 8;
  const float* qp = qW + ((long)(b * 64 + h * 8 + (q >> 3))) * 512 + (q & 7) * 64;
  for (int i = 0; i < 8; ++i) {
    int v = vb + i;
    const float* kp = kcW + ((long)(b * 32 + h * 4 + (v >> 3))) * 512 + (v & 7) * 64;
    float s = 0.f;
    for (int d = 0; d < 64; ++d) s += qp[d] * kp[d];
    sc[q][v] = 8.0f * s;
  }
  __syncthreads();
  if (t < 64) {
    float m = -3.4e38f;
    for (int v = 0; v < 32; ++v) m = fmaxf(m, sc[t][v]);
    float s = 0.f;
    for (int v = 0; v < 32; ++v) { float e = expf(sc[t][v] - m); sc[t][v] = e; s += e; }
    float inv = 1.0f / s;
    for (int v = 0; v < 32; ++v) sc[t][v] *= inv;
  }
  __syncthreads();
  if (t < 32) {
    float s = 0.f;
    for (int q2 = 0; q2 < 64; ++q2) s += sc[q2][t];
    wsum[(b * 8 + h) * 32 + t] = s;
  }
}

// ---------------------------------------------------------------------------
// Fused stage-1 value path. Per block (b,h,l2): 64x64 output, K = 8 v0-blocks
// of 512 e each. A-hat[l1, e] = sum_{v1<4} w[8v1+v0]*memory[b, h*64+l1, 4l2+v1, e]
// B-hat[e, d] = Wv_c[e, v0*64+d].  Reads its `memory` slice 8 times (once per v0).
// Writes mem_agg[b, 8*l1+l2, h*64+d].
// ---------------------------------------------------------------------------
__global__ __launch_bounds__(256) void stage1v_kernel(
    const float* __restrict__ mem, const float* __restrict__ Wv_c,
    const float* __restrict__ wsum, float* __restrict__ memAgg)
{
  __shared__ __align__(16) __bf16 Als[64 * 32];
  __shared__ __align__(16) __bf16 Bls[64 * 32];
  __shared__ float wsh[32];
  const int l2 = blockIdx.x, h = blockIdx.y, b = blockIdx.z;
  const int t = threadIdx.x;
  if (t < 32) wsh[t] = wsum[(b * 8 + h) * 32 + t];
  const int lane = t & 31, w = t >> 5;
  const int wm = w & 3, wn = w >> 2;
  const int half = lane >> 4, lr = lane & 15;
  v8f c0 = {}, c1 = {};

  const long memBase = (((long)(b * 512 + h * 64)) * 32 + 4 * l2) * 512;
  const int a_l1 = t >> 2, a_k0 = (t & 3) * 8;
  const int b_kk = t >> 3, b_n0 = (t & 7) * 8;

  for (int ks = 0; ks < 4096; ks += 32) {
    const int v0 = ks >> 9;      // which scale/column group of Wv_c
    const int e0 = ks & 511;     // e-offset within it
    __syncthreads();
    { // A-hat: weighted 4-row reduction of memory, f32 -> bf16 -> LDS
      const float* gp = mem + memBase + (long)a_l1 * 16384 + e0 + a_k0;
      float acc[8] = {0, 0, 0, 0, 0, 0, 0, 0};
#pragma unroll
      for (int v1 = 0; v1 < 4; ++v1) {
        float wv = wsh[8 * v1 + v0];
        float4 fa = *(const float4*)(gp + v1 * 512);
        float4 fb = *(const float4*)(gp + v1 * 512 + 4);
        acc[0] += wv * fa.x; acc[1] += wv * fa.y;
        acc[2] += wv * fa.z; acc[3] += wv * fa.w;
        acc[4] += wv * fb.x; acc[5] += wv * fb.y;
        acc[6] += wv * fb.z; acc[7] += wv * fb.w;
      }
      __builtin_prefetch(gp + 32, 0, 1);   // next k-step of the stream
      __bf16* d = &Als[a_l1 * 32 + a_k0];
#pragma unroll
      for (int i = 0; i < 8; ++i) d[i] = (__bf16)acc[i];
    }
    { // B-hat: permuted Wv_c column block (L2-resident)
      const float* gb = Wv_c + (long)(e0 + b_kk) * 512 + v0 * 64 + b_n0;
      float4 f0 = *(const float4*)gb;
      float4 f1 = *(const float4*)(gb + 4);
      float v[8] = {f0.x, f0.y, f0.z, f0.w, f1.x, f1.y, f1.z, f1.w};
#pragma unroll
      for (int i = 0; i < 8; ++i) Bls[(b_n0 + i) * 32 + b_kk] = (__bf16)v[i];
    }
    __syncthreads();
    v16bf a  = frag_a(&Als[(wm * 16 + lr) * 32], half);
    v16bf b0 = frag_b(&Bls[(wn * 32 + lr) * 32], half);
    v16bf b1 = frag_b(&Bls[(wn * 32 + 16 + lr) * 32], half);
    c0 = wmma_bf16(a, b0, c0);
    c1 = wmma_bf16(a, b1, c1);
  }
  const int l1base = wm * 16 + half * 8;
  const int d0 = wn * 32 + lr, d1 = d0 + 16;
#pragma unroll
  for (int r = 0; r < 8; ++r) {
    long row = (long)b * 512 + 8 * (l1base + r) + l2;
    float* cp = memAgg + row * 512 + h * 64;
    cp[d0] = c0[r];
    cp[d1] = c1[r];
  }
}

// ---------------------------------------------------------------------------
// Row softmax over 512 (stage-2 scores), one block per row.
// ---------------------------------------------------------------------------
__global__ __launch_bounds__(256) void softmax512_kernel(float* __restrict__ s) {
  __shared__ float red[256];
  float* p = s + (long)blockIdx.x * 512;
  const int t = threadIdx.x;
  float a = p[t], b = p[t + 256];
  red[t] = fmaxf(a, b);
  __syncthreads();
  for (int off = 128; off > 0; off >>= 1) {
    if (t < off) red[t] = fmaxf(red[t], red[t + off]);
    __syncthreads();
  }
  float m = red[0];
  __syncthreads();
  float e0 = expf(a - m), e1 = expf(b - m);
  red[t] = e0 + e1;
  __syncthreads();
  for (int off = 128; off > 0; off >>= 1) {
    if (t < off) red[t] += red[t + off];
    __syncthreads();
  }
  float inv = 1.0f / red[0];
  p[t] = e0 * inv;
  p[t + 256] = e1 * inv;
}

// ---------------------------------------------------------------------------
extern "C" void kernel_launch(void* const* d_in, const int* in_sizes, int n_in,
                              void* d_out, int out_size, void* d_ws, size_t ws_size,
                              hipStream_t stream) {
  (void)in_sizes; (void)n_in; (void)out_size; (void)ws_size;
  const float* query  = (const float*)d_in[0];
  const float* memory = (const float*)d_in[1];
  const float* Wq_c   = (const float*)d_in[2];
  const float* Wk_c   = (const float*)d_in[3];
  const float* Wv_c   = (const float*)d_in[4];
  const float* Wq     = (const float*)d_in[5];
  const float* Wk     = (const float*)d_in[6];
  const float* Wv     = (const float*)d_in[7];
  const float* Wp     = (const float*)d_in[8];
  const float* bp     = (const float*)d_in[9];
  float* out = (float*)d_out;
  float* ws  = (float*)d_ws;

  float* memMean = ws;                 // (B*V, 512)        131072
  float* qW      = memMean + 131072;   // (B*64, 512)       262144
  float* kcW     = qW + 262144;        // (B*V, 512)        131072
  float* wsum    = kcW + 131072;       // (B*H, 32)           2048
  float* memAgg  = wsum + 2048;        // (B*512, 512)     2097152
  float* Q2      = memAgg + 2097152;   // (B*64, 512)       262144
  float* Kmat    = Q2 + 262144;        // (B*512, 512)     2097152
  float* Vmat    = Kmat + 2097152;     // (B*512, 512)     2097152
  float* scores  = Vmat + 2097152;     // (B*H*64, 512)    2097152
  float* attnO   = scores + 2097152;   // (B*64, 512)       262144

  auto gemm = [&](const float* A, const float* Bm, float* C, const float* bias,
                  int M, int N, int K, int lda, int ldb, int ldc,
                  int transB, float alpha, int nb, int nh,
                  long sAb, long sAh, long sBb, long sBh, long sCb, long sCh) {
    dim3 g(N / 64, M / 64, nb * nh);
    gemm_wmma_kernel<<<g, 256, 0, stream>>>(A, Bm, C, bias, M, N, K, lda, ldb, ldc,
                                            transB, alpha, nh,
                                            sAb, sAh, sBb, sBh, sCb, sCh);
  };

  // Stage 1
  mean_kernel<<<512, 256, 0, stream>>>(memory, memMean);                 // pass 1 over memory
  gemm(query,   Wq_c, qW,  nullptr, 512, 512, 512, 512, 512, 512, 0, 1.f, 1, 1, 0,0,0,0,0,0);
  gemm(memMean, Wk_c, kcW, nullptr, 256, 512, 512, 512, 512, 512, 0, 1.f, 1, 1, 0,0,0,0,0,0);
  ch_softmax_kernel<<<64, 256, 0, stream>>>(qW, kcW, wsum);
  stage1v_kernel<<<dim3(8, 8, 8), 256, 0, stream>>>(memory, Wv_c, wsum, memAgg); // pass 2

  // Stage 2
  gemm(query,  Wq, Q2,   nullptr, 512,  512, 512, 512, 512, 512, 0, 1.f, 1, 1, 0,0,0,0,0,0);
  gemm(memAgg, Wk, Kmat, nullptr, 4096, 512, 512, 512, 512, 512, 0, 1.f, 1, 1, 0,0,0,0,0,0);
  gemm(memAgg, Wv, Vmat, nullptr, 4096, 512, 512, 512, 512, 512, 0, 1.f, 1, 1, 0,0,0,0,0,0);
  // scores[b,h] = 8 * Q_slice(64x64) @ K_slice^T(64x512)
  gemm(Q2, Kmat, scores, nullptr, 64, 512, 64, 512, 512, 512, 1, 8.f, 8, 8,
       32768, 64, 262144, 64, 262144, 32768);
  softmax512_kernel<<<4096, 256, 0, stream>>>(scores);
  // attn[b,h] = probs(64x512) @ V_slice(512x64) -> attnO[b, q, h*64+d]
  gemm(scores, Vmat, attnO, nullptr, 64, 64, 512, 512, 512, 512, 0, 1.f, 8, 8,
       262144, 32768, 262144, 64, 32768, 64);
  // out = attnO @ Wp + bp
  gemm(attnO, Wp, out, bp, 512, 512, 512, 512, 512, 512, 0, 1.f, 1, 1, 0,0,0,0,0,0);
}
